// SelfAttention_84490596647198
// MI455X (gfx1250) — compile-verified, matrix operand order checked by reading, not run on hardware
//
#include <hip/hip_runtime.h>
#include <hip/hip_bf16.h>
#include <stdint.h>

#define BATCH   4
#define SEQ     1024
#define DMODEL  1024
#define NH      16
#define HD      64
#define HIDDEN  1024
#define RTAB    255
#define RPAD    256

typedef __attribute__((ext_vector_type(16))) __bf16 v16bf;
typedef __attribute__((ext_vector_type(8)))  float  v8f;

union FragBF { uint4 q[2]; unsigned short u[16]; v16bf v; };

__device__ __forceinline__ unsigned short f2bf(float x) {
    unsigned int u = __float_as_uint(x);
    unsigned int r = u + 0x7FFFu + ((u >> 16) & 1u);   // round-to-nearest-even
    return (unsigned short)(r >> 16);
}

// Build a 16x32 bf16 fragment in the ISA A-layout (also used for B operands,
// which are fed from transposed sources so the load pattern is identical).
// lane L: row r = L%16, half hf = L/16; elements 0..7 <- K = hf*8+0..7,
// elements 8..15 <- K = 16+hf*8+0..7.  Two 16-byte loads per lane.
__device__ __forceinline__ v16bf frag_ld(const unsigned short* base, int stride_h, int lane) {
    int r = lane & 15, hf = lane >> 4;
    const unsigned short* p = base + (size_t)r * stride_h + hf * 8;
    FragBF f;
    f.q[0] = *(const uint4*)(p);
    f.q[1] = *(const uint4*)(p + 16);
    return f.v;
}

__device__ __forceinline__ v8f wmma_bf16(v16bf a, v16bf b, v8f c) {
    return __builtin_amdgcn_wmma_f32_16x16x32_bf16(false, a, false, b, (short)0, c, false, false);
}

// CDNA5 async copy: LDS[lds_off] <- MEM[sbase + goff], 16 bytes, ASYNCcnt-tracked.
__device__ __forceinline__ void async_cp16(unsigned lds_off, unsigned long long sbase, unsigned goff) {
    asm volatile("global_load_async_to_lds_b128 %0, %1, %2"
                 :: "v"(lds_off), "v"(goff), "s"(sbase)
                 : "memory");
}

// ---------------------------------------------------------------------------
// Kernel 1: qkv = x @ w_qkv + b_qkv, scattered to Qh/Kh [b,h,s,d] and Vt [b,h,d,s] (bf16)
// WG: 64x64 C-tile, 8 waves, K-loop step 32, LDS-staged A (bf16) and B^T (bf16).
// ---------------------------------------------------------------------------
__global__ __launch_bounds__(256) void qkv_gemm_kernel(
    const float* __restrict__ x, const float* __restrict__ w, const float* __restrict__ bias,
    unsigned short* __restrict__ Qh, unsigned short* __restrict__ Kh, unsigned short* __restrict__ Vt)
{
    __shared__ __align__(16) unsigned short As[64 * 32];   // [m][k]
    __shared__ __align__(16) unsigned short Bs[64 * 32];   // [n][k] (transposed)
    const int tid = threadIdx.x;
    const int lane = tid & 31, wave = tid >> 5;
    const int n0 = blockIdx.x * 64;
    const int m0 = blockIdx.y * 64;
    const int mi = (wave & 3) * 16;
    const int ni0 = (wave >> 2) * 16;          // wave owns n-subtiles ni0 and ni0+32
    v8f acc0 = {}, acc1 = {};

    for (int kb = 0; kb < DMODEL; kb += 32) {
        { // stage A 64x32 (f32 -> bf16)
            int row = tid >> 2;
            int col = (tid & 3) * 8;
            const float* src = x + (size_t)(m0 + row) * DMODEL + kb + col;
            if (kb + 32 < DMODEL) __builtin_prefetch(src + 32, 0, 1);   // global_prefetch_b8
            unsigned short* dst = As + row * 32 + col;
            #pragma unroll
            for (int i = 0; i < 8; ++i) dst[i] = f2bf(src[i]);
        }
        { // stage B 32x64 transposed: Bs[n][k] = w[kb+kk][n0+n]
            int kk = tid >> 3;
            int nn = (tid & 7) * 8;
            const float* src = w + (size_t)(kb + kk) * (3 * HIDDEN) + n0 + nn;
            if (kb + 32 < DMODEL) __builtin_prefetch(src + (size_t)32 * (3 * HIDDEN), 0, 1);
            #pragma unroll
            for (int i = 0; i < 8; ++i) Bs[(nn + i) * 32 + kk] = f2bf(src[i]);
        }
        __syncthreads();
        v16bf a  = frag_ld(As + mi * 32, 32, lane);
        v16bf b0 = frag_ld(Bs + ni0 * 32, 32, lane);
        v16bf b1 = frag_ld(Bs + (ni0 + 32) * 32, 32, lane);
        acc0 = wmma_bf16(a, b0, acc0);
        acc1 = wmma_bf16(a, b1, acc1);
        __syncthreads();
    }

    const int nl = lane & 15, hf = lane >> 4;
    #pragma unroll
    for (int e = 0; e < 8; ++e) {
        int m = m0 + mi + e + 8 * hf;
        int bb = m >> 10, s = m & 1023;
        #pragma unroll
        for (int t = 0; t < 2; ++t) {
            int c = n0 + ni0 + t * 32 + nl;
            float v = (t ? acc1[e] : acc0[e]) + bias[c];
            int which = c >> 10, wi = c & 1023;
            int d = wi >> 4, hh = wi & 15;        // hidden = d*16 + h
            unsigned short bv = f2bf(v);
            size_t bh = (size_t)bb * NH + hh;
            if (which == 0)      Qh[(bh * SEQ + s) * HD + d] = bv;
            else if (which == 1) Kh[(bh * SEQ + s) * HD + d] = bv;
            else                 Vt[(bh * HD + d) * SEQ + s] = bv;   // V transposed
        }
    }
}

// ---------------------------------------------------------------------------
// Kernel 2: Pr[b,h,q,r] = sum_d Qh[b,h,q,d] * rel_table[r,d]   (r padded to 256)
// ---------------------------------------------------------------------------
__global__ __launch_bounds__(256) void relproj_kernel(
    const float* __restrict__ rel_table, const unsigned short* __restrict__ Qh,
    float* __restrict__ Pr)
{
    __shared__ __align__(16) unsigned short Rs[RPAD * HD];  // [r][d] bf16
    const int tid = threadIdx.x, lane = tid & 31, wave = tid >> 5;
    const int b = blockIdx.z, h = blockIdx.y, q0 = blockIdx.x * 64;

    { // stage rel_table as bf16 (row 255 zero-padded)
        int rrow = tid;
        if (rrow < RTAB) {
            const float* src = rel_table + (size_t)rrow * HD;
            #pragma unroll 4
            for (int d = 0; d < HD; ++d) Rs[rrow * HD + d] = f2bf(src[d]);
        } else {
            for (int d = 0; d < HD; ++d) Rs[rrow * HD + d] = 0;
        }
    }
    __syncthreads();

    const size_t bh = (size_t)b * NH + h;
    const int mi = (wave & 3) * 16;
    const int nbase = (wave >> 2) * 128;
    const unsigned short* qbase = Qh + (bh * SEQ + q0 + mi) * HD;
    v16bf a0 = frag_ld(qbase, HD, lane);        // d 0..31
    v16bf a1 = frag_ld(qbase + 32, HD, lane);   // d 32..63
    const int nl = lane & 15, hf = lane >> 4;

    #pragma unroll
    for (int j = 0; j < 8; ++j) {
        int rn0 = nbase + j * 16;
        v16bf b0 = frag_ld(Rs + rn0 * HD, HD, lane);
        v16bf b1 = frag_ld(Rs + rn0 * HD + 32, HD, lane);
        v8f acc = {};
        acc = wmma_bf16(a0, b0, acc);
        acc = wmma_bf16(a1, b1, acc);
        #pragma unroll
        for (int e = 0; e < 8; ++e) {
            int q = q0 + mi + e + 8 * hf;
            Pr[(bh * SEQ + q) * RPAD + rn0 + nl] = acc[e];
        }
    }
}

// ---------------------------------------------------------------------------
// Kernel 3: flash attention. WG = (b, 32 q-rows), 16 waves = one head each
// (mask tile shared across heads via cache). k in chunks of 32, online softmax,
// P-tile transposed C-layout -> A-layout through a per-wave LDS buffer.
// K/V chunks are double-buffered in LDS via CDNA5 async-to-LDS copies so the
// next chunk streams in while WMMAs run on the current one.
// ---------------------------------------------------------------------------
__global__ __launch_bounds__(512) void attn_kernel(
    const unsigned short* __restrict__ Qh, const unsigned short* __restrict__ Kh,
    const unsigned short* __restrict__ Vt, const float* __restrict__ Pr,
    const int* __restrict__ mask, unsigned short* __restrict__ Ob)
{
    __shared__ __align__(16) unsigned short Ps[16 * 16 * 32];       // 16 waves x (16x32)   16 KB
    __shared__ __align__(16) unsigned short KbufS[2][16][32 * 64];  // dbl-buf K chunks    128 KB
    __shared__ __align__(16) unsigned short VbufS[2][16][64 * 32];  // dbl-buf V chunks    128 KB
    const int tid = threadIdx.x, lane = tid & 31;
    const int h = tid >> 5;                 // wave id == head
    const int b = blockIdx.y;
    const int q0 = blockIdx.x * 32;
    const size_t bh = (size_t)b * NH + h;
    const unsigned short* qb  = Qh + (bh * SEQ + q0) * HD;
    const unsigned short* kbp = Kh + bh * SEQ * HD;
    const unsigned short* vb  = Vt + bh * HD * SEQ;
    const float* prb = Pr + bh * SEQ * RPAD;
    unsigned short* pbuf = Ps + h * (16 * 32);
    const int nl = lane & 15, hf = lane >> 4;

    // LDS byte offsets of this wave's buffers (flat LDS aperture keeps the
    // wave-relative LDS offset in the low 32 address bits).
    unsigned kls[2] = { (unsigned)(size_t)&KbufS[0][h][0], (unsigned)(size_t)&KbufS[1][h][0] };
    unsigned vls[2] = { (unsigned)(size_t)&VbufS[0][h][0], (unsigned)(size_t)&VbufS[1][h][0] };

    // Stage one 32-wide K/V chunk: 16 async b128 per lane-set (4 KB K + 4 KB V).
    auto stage_kv = [&](int kk, int buf) {
        unsigned long long kbase = (unsigned long long)(kbp + (size_t)kk * HD); // 32x64 contiguous
        unsigned long long vbase = (unsigned long long)vb;
        #pragma unroll
        for (int i = 0; i < 4; ++i) {                       // K: 4096 B linear
            unsigned off = (unsigned)(i * 1024 + lane * 32);
            async_cp16(kls[buf] + off,      kbase, off);
            async_cp16(kls[buf] + off + 16, kbase, off + 16);
        }
        #pragma unroll
        for (int i = 0; i < 8; ++i) {                       // V: 64 rows x 64 B, stride SEQ
            int idx = i * 32 + lane;                        // 0..255
            int row = idx >> 2, part = idx & 3;
            unsigned goff = (unsigned)((row * SEQ + kk) * 2 + part * 16);
            unsigned loff = (unsigned)(row * 64 + part * 16);
            async_cp16(vls[buf] + loff, vbase, goff);
        }
    };

    v16bf qf[2][2];
    #pragma unroll
    for (int t = 0; t < 2; ++t) {
        qf[t][0] = frag_ld(qb + (size_t)(t * 16) * HD, HD, lane);
        qf[t][1] = frag_ld(qb + (size_t)(t * 16) * HD + 32, HD, lane);
    }
    v8f oacc[2][4];
    float rmax[2][8], rsum[2][8];
    #pragma unroll
    for (int t = 0; t < 2; ++t) {
        #pragma unroll
        for (int dt = 0; dt < 4; ++dt) oacc[t][dt] = (v8f){};
        #pragma unroll
        for (int e = 0; e < 8; ++e) { rmax[t][e] = -INFINITY; rsum[t][e] = 0.0f; }
    }

    stage_kv(0, 0);                                         // prologue fill

    for (int kk = 0; kk < SEQ; kk += 32) {
        int cur = (kk >> 5) & 1;
        if (kk + 32 < SEQ) {
            stage_kv(kk + 32, cur ^ 1);                     // prefetch next chunk
            // 16 async ops outstanding for next chunk; in-order completion
            // means ASYNCcnt<=16 <=> current chunk fully resident.
            asm volatile("s_wait_asynccnt 16" ::: "memory");
        } else {
            asm volatile("s_wait_asynccnt 0" ::: "memory");
        }
        const unsigned short* kl = &KbufS[cur][h][0];
        const unsigned short* vl = &VbufS[cur][h][0];

        v16bf kf[2][2];   // B-frags of K^T: n = k-col, contiguous = d
        #pragma unroll
        for (int j = 0; j < 2; ++j) {
            kf[j][0] = frag_ld(kl + j * 16 * 64, 64, lane);
            kf[j][1] = frag_ld(kl + j * 16 * 64 + 32, 64, lane);
        }
        v16bf vf[4];      // B-frags of V: n = d-col (rows of Vt), contiguous = k
        #pragma unroll
        for (int dt = 0; dt < 4; ++dt)
            vf[dt] = frag_ld(vl + dt * 16 * 32, 32, lane);

        #pragma unroll
        for (int t = 0; t < 2; ++t) {
            v8f s0 = {}, s1 = {};
            s0 = wmma_bf16(qf[t][0], kf[0][0], s0);
            s0 = wmma_bf16(qf[t][1], kf[0][1], s0);
            s1 = wmma_bf16(qf[t][0], kf[1][0], s1);
            s1 = wmma_bf16(qf[t][1], kf[1][1], s1);

            float sv0[8], sv1[8];
            #pragma unroll
            for (int e = 0; e < 8; ++e) {
                int q = q0 + t * 16 + e + 8 * hf;
                int k0i = kk + nl, k1i = kk + 16 + nl;
                int i0 = min(max(k0i - q, -127), 127) + 127;
                int i1 = min(max(k1i - q, -127), 127) + 127;
                float x0 = 8.0f * s0[e] + prb[(size_t)q * RPAD + i0];  // scale=sqrt(64)
                float x1 = 8.0f * s1[e] + prb[(size_t)q * RPAD + i1];
                const int* mrow = mask + ((size_t)b * SEQ + q) * SEQ;
                if (mrow[k0i] == 0) x0 = -1e8f;
                if (mrow[k1i] == 0) x1 = -1e8f;
                sv0[e] = x0; sv1[e] = x1;
            }
            // online softmax per row (row stats broadcast within 16-lane group)
            #pragma unroll
            for (int e = 0; e < 8; ++e) {
                float v = fmaxf(sv0[e], sv1[e]);
                #pragma unroll
                for (int m = 1; m <= 8; m <<= 1) v = fmaxf(v, __shfl_xor(v, m, 32));
                float mnew  = fmaxf(rmax[t][e], v);
                float scale = __expf(rmax[t][e] - mnew);     // exp(-inf)=0 first chunk
                float p0 = __expf(sv0[e] - mnew);
                float p1 = __expf(sv1[e] - mnew);
                float ps = p0 + p1;
                #pragma unroll
                for (int m = 1; m <= 8; m <<= 1) ps += __shfl_xor(ps, m, 32);
                rsum[t][e] = rsum[t][e] * scale + ps;
                rmax[t][e] = mnew;
                sv0[e] = p0; sv1[e] = p1;
                #pragma unroll
                for (int dt = 0; dt < 4; ++dt) oacc[t][dt][e] *= scale;
            }
            // C-layout -> A-layout transpose of P through per-wave LDS slice
            #pragma unroll
            for (int e = 0; e < 8; ++e) {
                int m = e + 8 * hf;
                pbuf[m * 32 + nl]      = f2bf(sv0[e]);
                pbuf[m * 32 + 16 + nl] = f2bf(sv1[e]);
            }
            v16bf pf = frag_ld(pbuf, 32, lane);   // same-wave RAW: dscnt wait by compiler
            #pragma unroll
            for (int dt = 0; dt < 4; ++dt)
                oacc[t][dt] = wmma_bf16(pf, vf[dt], oacc[t][dt]);
        }
    }

    #pragma unroll
    for (int t = 0; t < 2; ++t) {
        #pragma unroll
        for (int e = 0; e < 8; ++e) {
            float inv = 1.0f / rsum[t][e];
            int q = q0 + t * 16 + e + 8 * hf;
            #pragma unroll
            for (int dt = 0; dt < 4; ++dt) {
                int d = dt * 16 + nl;
                Ob[((size_t)b * SEQ + q) * HIDDEN + d * NH + h] = f2bf(oacc[t][dt][e] * inv);
            }
        }
    }
}

// ---------------------------------------------------------------------------
// Kernel 4: out = Ob @ w_out + b_out  (A direct-from-global bf16, B staged transposed)
// ---------------------------------------------------------------------------
__global__ __launch_bounds__(256) void out_gemm_kernel(
    const unsigned short* __restrict__ Ob, const float* __restrict__ w_out,
    const float* __restrict__ b_out, float* __restrict__ out)
{
    __shared__ __align__(16) unsigned short Bs[64 * 32];   // [n][k]
    const int tid = threadIdx.x, lane = tid & 31, wave = tid >> 5;
    const int n0 = blockIdx.x * 64, m0 = blockIdx.y * 64;
    const int mi = (wave & 3) * 16, ni0 = (wave >> 2) * 16;
    v8f acc0 = {}, acc1 = {};

    for (int kb = 0; kb < HIDDEN; kb += 32) {
        {
            int kk = tid >> 3;
            int nn = (tid & 7) * 8;
            const float* src = w_out + (size_t)(kb + kk) * DMODEL + n0 + nn;
            if (kb + 32 < HIDDEN) __builtin_prefetch(src + (size_t)32 * DMODEL, 0, 1);
            #pragma unroll
            for (int i = 0; i < 8; ++i) Bs[(nn + i) * 32 + kk] = f2bf(src[i]);
        }
        __syncthreads();
        v16bf a  = frag_ld(Ob + (size_t)(m0 + mi) * HIDDEN + kb, HIDDEN, lane);
        v16bf b0 = frag_ld(Bs + ni0 * 32, 32, lane);
        v16bf b1 = frag_ld(Bs + (ni0 + 32) * 32, 32, lane);
        acc0 = wmma_bf16(a, b0, acc0);
        acc1 = wmma_bf16(a, b1, acc1);
        __syncthreads();
    }

    const int nl = lane & 15, hf = lane >> 4;
    #pragma unroll
    for (int e = 0; e < 8; ++e) {
        int m = m0 + mi + e + 8 * hf;
        int c0 = n0 + ni0 + nl, c1 = c0 + 32;
        out[(size_t)m * DMODEL + c0] = acc0[e] + b_out[c0];
        out[(size_t)m * DMODEL + c1] = acc1[e] + b_out[c1];
    }
}

// ---------------------------------------------------------------------------
extern "C" void kernel_launch(void* const* d_in, const int* in_sizes, int n_in,
                              void* d_out, int out_size, void* d_ws, size_t ws_size,
                              hipStream_t stream)
{
    (void)in_sizes; (void)n_in; (void)out_size; (void)ws_size;
    const float* x         = (const float*)d_in[0];
    const int*   mask      = (const int*)  d_in[1];
    const float* w_qkv     = (const float*)d_in[2];
    const float* b_qkv     = (const float*)d_in[3];
    const float* rel_table = (const float*)d_in[4];
    const float* w_out     = (const float*)d_in[5];
    const float* b_out     = (const float*)d_in[6];
    float* out = (float*)d_out;

    char* ws = (char*)d_ws;
    size_t off = 0;
    auto alloc = [&](size_t bytes) -> void* {
        void* p = ws + off;
        off = (off + bytes + 255) & ~(size_t)255;
        return p;
    };
    const size_t nbhsd = (size_t)BATCH * NH * SEQ * HD;           // 4,194,304
    unsigned short* Qh = (unsigned short*)alloc(nbhsd * 2);       // 8 MB
    unsigned short* Kh = (unsigned short*)alloc(nbhsd * 2);       // 8 MB
    unsigned short* Vt = (unsigned short*)alloc(nbhsd * 2);       // 8 MB (transposed)
    float* Pr = (float*)alloc((size_t)BATCH * NH * SEQ * RPAD * 4); // 64 MB
    unsigned short* Ob = (unsigned short*)alloc((size_t)BATCH * SEQ * HIDDEN * 2); // 8 MB

    dim3 g1(3 * HIDDEN / 64, (BATCH * SEQ) / 64);   // 48 x 64
    qkv_gemm_kernel<<<g1, 256, 0, stream>>>(x, w_qkv, b_qkv, Qh, Kh, Vt);

    dim3 g2(SEQ / 64, NH, BATCH);                   // 16 x 16 x 4
    relproj_kernel<<<g2, 256, 0, stream>>>(rel_table, Qh, Pr);

    dim3 g3(SEQ / 32, BATCH);                       // 32 x 4, 512 threads (16 waves)
    attn_kernel<<<g3, 512, 0, stream>>>(Qh, Kh, Vt, Pr, mask, Ob);

    dim3 g4(DMODEL / 64, (BATCH * SEQ) / 64);       // 16 x 64
    out_gemm_kernel<<<g4, 256, 0, stream>>>(Ob, w_out, b_out, out);
}
